// ResidualBinaryLayer_49168785605336
// MI455X (gfx1250) — 8-bit matrix operand order repaired and checked by reading, compiled, not run on hardware
//
#include <hip/hip_runtime.h>

typedef int v8i __attribute__((ext_vector_type(8)));

#define N_   16
#define C_   64
#define H_   224
#define W_   224
#define O_   64
#define HW_  (H_ * W_)
#define NHW_ (N_ * HW_)
#define HP_  226
#define WP_  226
#define PIX_PER_N_ (HP_ * WP_)
#define NPIX_ (N_ * PIX_PER_N_)
#define TPR_ 14

#define WS_STATS 0
#define WS_BFRAG 1024
#define WS_APAD  40960

__global__ void rbl_zero_stats(float* __restrict__ stats) {
    if (threadIdx.x < 128) stats[threadIdx.x] = 0.0f;
}

__global__ __launch_bounds__(256)
void rbl_stats(const float* __restrict__ x, float* __restrict__ stats) {
    const int c = blockIdx.x;
    const int n = blockIdx.y;
    const float* __restrict__ p = x + (size_t)(n * C_ + c) * HW_;
    float s = 0.f, ss = 0.f;
    for (int i = threadIdx.x; i < HW_; i += 256) {
        float v = p[i];
        s += v; ss += v * v;
    }
    __shared__ float red0[256];
    __shared__ float red1[256];
    red0[threadIdx.x] = s; red1[threadIdx.x] = ss;
    __syncthreads();
    for (int off = 128; off > 0; off >>= 1) {
        if ((int)threadIdx.x < off) {
            red0[threadIdx.x] += red0[threadIdx.x + off];
            red1[threadIdx.x] += red1[threadIdx.x + off];
        }
        __syncthreads();
    }
    if (threadIdx.x == 0) {
        atomicAdd(&stats[c], red0[0]);
        atomicAdd(&stats[C_ + c], red1[0]);
    }
}

__global__ void rbl_finalize(float* __restrict__ stats,
                             const float* __restrict__ gamma,
                             const float* __restrict__ beta) {
    int c = threadIdx.x;
    if (c >= C_) return;
    const float inv = 1.0f / (float)NHW_;
    float m   = stats[c] * inv;
    float var = stats[C_ + c] * inv - m * m;
    float sc  = gamma[c] * rsqrtf(var + 1e-5f);
    stats[128 + c] = sc;
    stats[192 + c] = beta[c] - m * sc;
}

__global__ __launch_bounds__(256)
void rbl_packw(const float* __restrict__ weight, int* __restrict__ bfrag) {
    int d = blockIdx.x * 256 + threadIdx.x;
    if (d >= 9216) return;
    int v  = d & 7;
    int l  = (d >> 3) & 31;
    int nt = (d >> 8) & 3;
    int t  = d >> 10;
    int o  = nt * 16 + (l & 15);
    int kh = t / 3, kw = t % 3;
    unsigned int acc = 0u;
    #pragma unroll
    for (int j = 0; j < 4; ++j) {
        int c = ((v >> 1) * 16) + ((l >> 4) * 8) + ((v & 1) * 4) + j;
        float wv = weight[(((size_t)o * C_ + c) * 3 + kh) * 3 + kw];
        unsigned int b = (wv >= 0.f) ? 0x01u : 0xFFu;
        acc |= b << (8 * j);
    }
    bfrag[d] = (int)acc;
}

__global__ __launch_bounds__(256)
void rbl_binarize(const float* __restrict__ x,
                  const float* __restrict__ stats,
                  signed char* __restrict__ apad) {
    __shared__ float sb[128];
    if (threadIdx.x < 128) sb[threadIdx.x] = stats[128 + threadIdx.x];
    __syncthreads();

    int pid = blockIdx.x * 256 + threadIdx.x;
    if (pid >= NPIX_) return;
    int n   = pid / PIX_PER_N_;
    int rem = pid % PIX_PER_N_;
    int hp  = rem / WP_, wp = rem % WP_;
    int4* dst = (int4*)(apad + (size_t)pid * 64);
    if (hp == 0 || hp == HP_ - 1 || wp == 0 || wp == WP_ - 1) {
        int4 z = make_int4(0, 0, 0, 0);
        dst[0] = z; dst[1] = z; dst[2] = z; dst[3] = z;
        return;
    }
    int h = hp - 1, w = wp - 1;
    unsigned int words[16];
    #pragma unroll 4
    for (int cw = 0; cw < 16; ++cw) {
        unsigned int acc = 0u;
        #pragma unroll
        for (int j = 0; j < 4; ++j) {
            int c = cw * 4 + j;
            float v = x[(size_t)(n * C_ + c) * HW_ + h * W_ + w];
            float t = v * sb[c] + sb[64 + c];
            unsigned int b = (t >= 0.f) ? 0x01u : 0xFFu;
            acc |= b << (8 * j);
        }
        words[cw] = acc;
    }
    #pragma unroll
    for (int q = 0; q < 4; ++q)
        dst[q] = make_int4((int)words[4*q], (int)words[4*q+1],
                           (int)words[4*q+2], (int)words[4*q+3]);
}


__device__ __forceinline__ void rbl_load_a(v8i a[9], const signed char* __restrict__ apad,
                                           int n, int h, int w0, int m, int hi8) {
    #pragma unroll
    for (int t = 0; t < 9; ++t) {
        const int kh = t / 3, kw = t - kh * 3;
        const int* pi = (const int*)(apad
            + ((size_t)((n * HP_ + h + kh) * WP_) + (w0 + m + kw)) * 64 + hi8);
        #pragma unroll
        for (int v = 0; v < 4; ++v) {
            a[t][2 * v + 0] = pi[v * 4 + 0];
            a[t][2 * v + 1] = pi[v * 4 + 1];
        }
    }
}

__device__ __forceinline__ void rbl_tile(const v8i a[9], const v8i b[9], float al,
                                         const float* __restrict__ x,
                                         float* __restrict__ out,
                                         size_t rowbase, int w0, int hi8) {
    v8i acc = (v8i){0, 0, 0, 0, 0, 0, 0, 0};
    #pragma unroll
    for (int t = 0; t < 9; ++t) {
        acc = __builtin_amdgcn_wmma_i32_16x16x64_iu8(true, a[t], true, b[t], acc,
                                                     false, false);
    }
    const size_t base = rowbase + (size_t)(w0 + hi8);
    #pragma unroll
    for (int r = 0; r < 8; ++r) {
        out[base + r] = al * (float)acc[r] + x[base + r];
    }
}

__global__ __launch_bounds__(128)
void rbl_gemm(const signed char* __restrict__ apad,
              const int* __restrict__ bfrag,
              const float* __restrict__ alpha,
              const float* __restrict__ x,
              float* __restrict__ out) {
    const int nt   = threadIdx.x >> 5;
    const int lane = threadIdx.x & 31;
    const int n = blockIdx.x / H_;
    const int h = blockIdx.x % H_;
    const int m   = lane & 15;
    const int hi8 = (lane >> 4) * 8;

    v8i b[9];
    #pragma unroll
    for (int t = 0; t < 9; ++t) {
        const int* bp = bfrag + ((size_t)((t * 4 + nt) * 32 + lane) * 8);
        #pragma unroll
        for (int v = 0; v < 8; ++v) b[t][v] = bp[v];
    }

    const int   o  = nt * 16 + (lane & 15);
    const float al = alpha[o];
    const size_t rowbase = ((size_t)(n * O_ + o) * H_ + h) * W_;

    v8i a0[9], a1[9];
    rbl_load_a(a0, apad, n, h, 0, m, hi8);
    __builtin_prefetch(x + rowbase + hi8, 0, 1);

    #pragma unroll 1
    for (int wt = 0; wt < TPR_; wt += 2) {
        rbl_load_a(a1, apad, n, h, (wt + 1) * 16, m, hi8);
        __builtin_prefetch(x + rowbase + (size_t)((wt + 1) * 16 + hi8), 0, 1);
        rbl_tile(a0, b, al, x, out, rowbase, wt * 16, hi8);

        if (wt + 2 < TPR_) {
            rbl_load_a(a0, apad, n, h, (wt + 2) * 16, m, hi8);
            __builtin_prefetch(x + rowbase + (size_t)((wt + 2) * 16 + hi8), 0, 1);
        }
        rbl_tile(a1, b, al, x, out, rowbase, (wt + 1) * 16, hi8);
    }
}

extern "C" void kernel_launch(void* const* d_in, const int* in_sizes, int n_in,
                              void* d_out, int out_size, void* d_ws, size_t ws_size,
                              hipStream_t stream) {
    const float* x      = (const float*)d_in[0];
    const float* gamma  = (const float*)d_in[1];
    const float* beta   = (const float*)d_in[2];
    const float* weight = (const float*)d_in[3];
    const float* alpha  = (const float*)d_in[4];
    float* out = (float*)d_out;

    float*       stats = (float*)((char*)d_ws + WS_STATS);
    int*         bfrag = (int*)((char*)d_ws + WS_BFRAG);
    signed char* apad  = (signed char*)((char*)d_ws + WS_APAD);

    rbl_zero_stats<<<1, 128, 0, stream>>>(stats);
    rbl_stats<<<dim3(C_, N_), 256, 0, stream>>>(x, stats);
    rbl_finalize<<<1, 64, 0, stream>>>(stats, gamma, beta);
    rbl_packw<<<36, 256, 0, stream>>>(weight, bfrag);
    rbl_binarize<<<(NPIX_ + 255) / 256, 256, 0, stream>>>(x, stats, apad);
    rbl_gemm<<<N_ * H_, 128, 0, stream>>>(apad, bfrag, alpha, x, out);
}
